// Attention_23373212025106
// MI455X (gfx1250) — matrix operand order repaired and checked by reading, not compiled, not run on hardware
//
#include <hip/hip_runtime.h>
#include <hip/hip_bf16.h>

typedef __bf16 bf16;
typedef __attribute__((ext_vector_type(16))) __bf16 v16bf;
typedef __attribute__((ext_vector_type(8)))  float  v8f;

union AFrag { v16bf v; unsigned u[8]; };
union BFrag { v16bf v; unsigned u[8]; };

__device__ __forceinline__ unsigned pack_bf16(float a, float b) {
  union { bf16 h[2]; unsigned u; } p;
  p.h[0] = (bf16)a; p.h[1] = (bf16)b;
  return p.u;
}

// A-matrix 16x32 bf16 fragment from LDS tile (row-major, `stride` halves).
// ISA 7.12.2: lanes 0-15 rows M; VGPR i holds K = (i<4?0:16)+half*8+(i%4)*2, +1
__device__ __forceinline__ void load_a(AFrag& f, const bf16* lds, int stride) {
  int lane = threadIdx.x & 31;
  const bf16* row = lds + (lane & 15) * stride;
  int half = lane >> 4;
#pragma unroll
  for (int i = 0; i < 8; ++i) {
    int k = ((i & 3) << 1) + (half << 3) + ((i >> 2) << 4);
    f.u[i] = *(const unsigned*)(row + k);
  }
}

// B-matrix 32x16 bf16 fragment. LDS holds B transposed: ldsT[n][k] (k contiguous).
// VGPR i holds K = (i<4?0:16)+half*8+(i%4)*2, +1 (same as A).
__device__ __forceinline__ void load_b(BFrag& f, const bf16* ldsT, int stride, int koff) {
  int lane = threadIdx.x & 31;
  const bf16* col = ldsT + (lane & 15) * stride + koff + ((lane >> 4) << 3);
#pragma unroll
  for (int i = 0; i < 8; ++i)
    f.u[i] = *(const unsigned*)(col + ((i & 3) << 1) + ((i >> 2) << 4));
}

__device__ __forceinline__ v8f wmma_bf16(const AFrag& a, const BFrag& b, v8f c) {
  return __builtin_amdgcn_wmma_f32_16x16x32_bf16(false, a.v, false, b.v, (short)0, c,
                                                 false, false);
}

// ---------------------------------------------------------------------------
// Kernel 1: QKV GEMM.  X[4096,1024] f32 @ W[1024,3072] f32 + b -> Q,K,V bf16.
// Block: 256 thr = 8 waves (4x2), tile 128(M) x 128(N); wave tile 32x64
// (2x4 = 8 WMMA per 32-deep k-step).  Q gets 1/sqrt(hd)=0.125 folded in.
// Outputs in [B,H,S,hd] bf16.
// ---------------------------------------------------------------------------
__global__ __launch_bounds__(256) void qkv_kernel(const float* __restrict__ X,
                                                  const float* __restrict__ W,
                                                  const float* __restrict__ bias,
                                                  bf16* __restrict__ Qw,
                                                  bf16* __restrict__ Kw,
                                                  bf16* __restrict__ Vw) {
  const int K = 1024, N = 3072;
  __shared__ bf16 As[128][34];   // [m][k]
  __shared__ bf16 Bt[128][34];   // [n][k] (transposed)
  int m0 = blockIdx.y * 128;
  int n0 = blockIdx.x * 128;
  int t = threadIdx.x;
  int wave = t >> 5, lane = t & 31;
  int wm = (wave >> 1) * 32;     // 4 waves along M
  int wn = (wave & 1) * 64;      // 2 waves along N
  v8f acc[2][4] = {};

  for (int k0 = 0; k0 < K; k0 += 32) {
    {  // stage A: thread covers 16 contiguous k of one row, packed b32 stores
      int row = t >> 1, ks = (t & 1) * 16;
      const float* src = X + (size_t)(m0 + row) * K + k0 + ks;
#pragma unroll
      for (int i = 0; i < 8; ++i)
        *(unsigned*)&As[row][ks + 2 * i] = pack_bf16(src[2 * i], src[2 * i + 1]);
    }
    {  // stage B transposed: thread reads 16 contiguous n of one k-row
      int kk = t >> 3, ns = (t & 7) * 16;
      const float* src = W + (size_t)(k0 + kk) * N + n0 + ns;
#pragma unroll
      for (int i = 0; i < 16; ++i) Bt[ns + i][kk] = (bf16)src[i];
    }
    if (k0 + 32 < K) {
      __builtin_prefetch(X + (size_t)(m0 + (t >> 1)) * K + k0 + 32, 0, 0);
      __builtin_prefetch(W + (size_t)(k0 + 32 + (t >> 3)) * N + n0 + (t & 7) * 16, 0, 0);
    }
    __syncthreads();
    AFrag a[2]; BFrag b[4];
    load_a(a[0], &As[wm][0], 34);
    load_a(a[1], &As[wm + 16][0], 34);
#pragma unroll
    for (int j = 0; j < 4; ++j) load_b(b[j], &Bt[wn + j * 16][0], 34, 0);
#pragma unroll
    for (int i = 0; i < 2; ++i)
#pragma unroll
      for (int j = 0; j < 4; ++j) acc[i][j] = wmma_bf16(a[i], b[j], acc[i][j]);
    __syncthreads();
  }

  // Epilogue: scatter into Q/K/V [B,H,S,hd] bf16 (+bias, Q scaled by 0.125)
  int nl = lane & 15, half = lane >> 4;
#pragma unroll
  for (int i = 0; i < 2; ++i)
#pragma unroll
    for (int j = 0; j < 4; ++j) {
      int gn = n0 + wn + j * 16 + nl;
      float bv = bias[gn];
      int region = gn >> 10;          // 0=q 1=k 2=v
      int d = gn & 1023;
      int h = d >> 6, dd = d & 63;
      bf16* dst = (region == 0) ? Qw : (region == 1) ? Kw : Vw;
      float scl = (region == 0) ? 0.125f : 1.0f;
#pragma unroll
      for (int r = 0; r < 8; ++r) {
        int gm = m0 + wm + i * 16 + half * 8 + r;
        int bb = gm >> 11, s = gm & 2047;
        size_t idx = ((size_t)((bb * 16 + h) * 2048 + s)) * 64 + dd;
        dst[idx] = (bf16)((acc[i][j][r] + bv) * scl);
      }
    }
}

// ---------------------------------------------------------------------------
// Kernel 2: causal flash attention.  Block = 4 waves (128 thr), 64 q-rows per
// block, each wave owns 16 q-rows (no cross-wave softmax).  Key blocks of 32
// keys so that P(16x32) @ V(32x64) is exactly one bf16-WMMA k-step.
// ---------------------------------------------------------------------------
__global__ __launch_bounds__(128) void attn_kernel(const bf16* __restrict__ Q,
                                                   const bf16* __restrict__ Kg,
                                                   const bf16* __restrict__ Vg,
                                                   bf16* __restrict__ Og) {
  __shared__ bf16 Qs[64][66];       // [q][hd]
  __shared__ bf16 Ks[32][66];       // [key][hd]  (== K^T columns for B-frags)
  __shared__ bf16 Vt[64][34];       // [hd][key]  (transposed V for B-frags)
  __shared__ bf16 Ps[4][16][32];    // wave-private P tiles

  int qt = blockIdx.x, bh = blockIdx.y;
  const bf16* Qb = Q + (size_t)bh * 2048 * 64;
  const bf16* Kb = Kg + (size_t)bh * 2048 * 64;
  const bf16* Vb = Vg + (size_t)bh * 2048 * 64;

  int t = threadIdx.x, wave = t >> 5, lane = t & 31;
  int nl = lane & 15, half = lane >> 4;

  {  // stage Q tile once (dword copies)
    int row = t >> 1, cs = (t & 1) * 32;
    const bf16* src = Qb + (size_t)(qt * 64 + row) * 64 + cs;
#pragma unroll
    for (int i = 0; i < 16; ++i)
      *(unsigned*)&Qs[row][cs + 2 * i] = *(const unsigned*)(src + 2 * i);
  }
  __syncthreads();

  AFrag aq[2];
  load_a(aq[0], &Qs[wave * 16][0], 66);    // hd 0..31
  load_a(aq[1], &Qs[wave * 16][32], 66);   // hd 32..63

  v8f o[4] = {};
  float mrow[8], lrow[8];
#pragma unroll
  for (int j = 0; j < 8; ++j) { mrow[j] = -1e30f; lrow[j] = 0.f; }

  int nkb = 2 * qt + 2;  // causal: keys <= last q row of this tile
  for (int kb = 0; kb < nkb; ++kb) {
    __syncthreads();  // prior iteration done reading Ks/Vt
    {                 // cooperative K/V stage (V transposed)
      int key = t >> 2, cs = (t & 3) * 16;
      const bf16* sk = Kb + (size_t)(kb * 32 + key) * 64 + cs;
      const bf16* sv = Vb + (size_t)(kb * 32 + key) * 64 + cs;
#pragma unroll
      for (int i = 0; i < 8; ++i)
        *(unsigned*)&Ks[key][cs + 2 * i] = *(const unsigned*)(sk + 2 * i);
#pragma unroll
      for (int i = 0; i < 16; ++i) Vt[cs + i][key] = sv[i];
    }
    __syncthreads();

    // scores: two 16x16 tiles, hd=64 -> 2 k-steps each
    v8f s[2] = {};
#pragma unroll
    for (int ts = 0; ts < 2; ++ts) {
      BFrag bk;
      load_b(bk, &Ks[ts * 16][0], 66, 0);
      s[ts] = wmma_bf16(aq[0], bk, s[ts]);
      load_b(bk, &Ks[ts * 16][0], 66, 32);
      s[ts] = wmma_bf16(aq[1], bk, s[ts]);
    }

    // causal mask
    int qbase = qt * 64 + wave * 16 + half * 8;
#pragma unroll
    for (int ts = 0; ts < 2; ++ts) {
      int kpos = kb * 32 + ts * 16 + nl;
#pragma unroll
      for (int j = 0; j < 8; ++j)
        if (kpos > qbase + j) s[ts][j] = -1e30f;
    }

    // online softmax (row reductions across the 16 lanes of each half)
    float p0[8], p1[8];
#pragma unroll
    for (int j = 0; j < 8; ++j) {
      float r = fmaxf(s[0][j], s[1][j]);
#pragma unroll
      for (int off = 1; off < 16; off <<= 1) r = fmaxf(r, __shfl_xor(r, off, 32));
      float mn = fmaxf(mrow[j], r);
      float sc = __expf(mrow[j] - mn);
      float ea = __expf(s[0][j] - mn);
      float eb = __expf(s[1][j] - mn);
      float rs = ea + eb;
#pragma unroll
      for (int off = 1; off < 16; off <<= 1) rs += __shfl_xor(rs, off, 32);
      lrow[j] = lrow[j] * sc + rs;
      mrow[j] = mn;
      p0[j] = ea; p1[j] = eb;
#pragma unroll
      for (int nt = 0; nt < 4; ++nt) o[nt][j] *= sc;
    }

    // C-layout -> A-layout via wave-private LDS
#pragma unroll
    for (int j = 0; j < 8; ++j) {
      int m = half * 8 + j;
      Ps[wave][m][nl] = (bf16)p0[j];
      Ps[wave][m][16 + nl] = (bf16)p1[j];
    }
    AFrag ap;
    load_a(ap, &Ps[wave][0][0], 32);
#pragma unroll
    for (int nt = 0; nt < 4; ++nt) {
      BFrag bv;
      load_b(bv, &Vt[nt * 16][0], 34, 0);
      o[nt] = wmma_bf16(ap, bv, o[nt]);
    }
  }

  // normalize and write attn output as bf16 [B,S,D] (heads re-interleaved)
  int b = bh >> 4, h = bh & 15;
  int qbase = qt * 64 + wave * 16 + half * 8;
#pragma unroll
  for (int j = 0; j < 8; ++j) {
    float inv = 1.0f / lrow[j];
    size_t base = ((size_t)(b * 2048 + qbase + j)) * 1024 + h * 64;
#pragma unroll
    for (int nt = 0; nt < 4; ++nt)
      Og[base + nt * 16 + nl] = (bf16)(o[nt][j] * inv);
  }
}

// ---------------------------------------------------------------------------
// Kernel 3: output projection.  A bf16 [4096,1024] @ Wp f32 [1024,1024] + bp
// -> f32 out.  Same 128x128 tiling as kernel 1.
// ---------------------------------------------------------------------------
__global__ __launch_bounds__(256) void proj_kernel(const bf16* __restrict__ A,
                                                   const float* __restrict__ W,
                                                   const float* __restrict__ bias,
                                                   float* __restrict__ out) {
  const int K = 1024, N = 1024;
  __shared__ bf16 As[128][34];
  __shared__ bf16 Bt[128][34];
  int m0 = blockIdx.y * 128;
  int n0 = blockIdx.x * 128;
  int t = threadIdx.x;
  int wave = t >> 5, lane = t & 31;
  int wm = (wave >> 1) * 32;
  int wn = (wave & 1) * 64;
  v8f acc[2][4] = {};

  for (int k0 = 0; k0 < K; k0 += 32) {
    {  // stage A (already bf16): dword copies
      int row = t >> 1, ks = (t & 1) * 16;
      const bf16* src = A + (size_t)(m0 + row) * K + k0 + ks;
#pragma unroll
      for (int i = 0; i < 8; ++i)
        *(unsigned*)&As[row][ks + 2 * i] = *(const unsigned*)(src + 2 * i);
    }
    {
      int kk = t >> 3, ns = (t & 7) * 16;
      const float* src = W + (size_t)(k0 + kk) * N + n0 + ns;
#pragma unroll
      for (int i = 0; i < 16; ++i) Bt[ns + i][kk] = (bf16)src[i];
    }
    __syncthreads();
    AFrag a[2]; BFrag b[4];
    load_a(a[0], &As[wm][0], 34);
    load_a(a[1], &As[wm + 16][0], 34);
#pragma unroll
    for (int j = 0; j < 4; ++j) load_b(b[j], &Bt[wn + j * 16][0], 34, 0);
#pragma unroll
    for (int i = 0; i < 2; ++i)
#pragma unroll
      for (int j = 0; j < 4; ++j) acc[i][j] = wmma_bf16(a[i], b[j], acc[i][j]);
    __syncthreads();
  }

  int nl = lane & 15, half = lane >> 4;
#pragma unroll
  for (int i = 0; i < 2; ++i)
#pragma unroll
    for (int j = 0; j < 4; ++j) {
      int gn = n0 + wn + j * 16 + nl;
      float bv = bias[gn];
#pragma unroll
      for (int r = 0; r < 8; ++r) {
        int gm = m0 + wm + i * 16 + half * 8 + r;
        out[(size_t)gm * N + gn] = acc[i][j][r] + bv;
      }
    }
}

// ---------------------------------------------------------------------------
extern "C" void kernel_launch(void* const* d_in, const int* in_sizes, int n_in,
                              void* d_out, int out_size, void* d_ws, size_t ws_size,
                              hipStream_t stream) {
  const float* X  = (const float*)d_in[0];  // hidden_states [2,2048,1024]
  const float* Wa = (const float*)d_in[1];  // w_attn [1024,3072]
  const float* ba = (const float*)d_in[2];  // b_attn [3072]
  const float* Wp = (const float*)d_in[3];  // w_proj [1024,1024]
  const float* bp = (const float*)d_in[4];  // b_proj [1024]
  float* out = (float*)d_out;

  const size_t per = (size_t)2 * 16 * 2048 * 64;  // 4M elems per tensor
  bf16* Qw = (bf16*)d_ws;
  bf16* Kw = Qw + per;
  bf16* Vw = Kw + per;
  bf16* Aw = Vw + per;  // attn output [B,S,D] bf16

  qkv_kernel<<<dim3(3072 / 128, 4096 / 128), 256, 0, stream>>>(X, Wa, ba, Qw, Kw, Vw);
  attn_kernel<<<dim3(2048 / 64, 2 * 16), 128, 0, stream>>>(Qw, Kw, Vw, Aw);
  proj_kernel<<<dim3(1024 / 128, 4096 / 128), 256, 0, stream>>>(Aw, Wp, bp, out);
}
